// Net_9440338117233
// MI455X (gfx1250) — matrix operand order repaired and checked by reading, not compiled, not run on hardware
//
#include <hip/hip_runtime.h>
#include <hip/hip_bf16.h>

// MI455X / gfx1250, wave32. Heavy layers via v_wmma_f32_16x16x32_bf16 (f32 accum).
// Fragment I/O is vectorized: A/B frags = 2 x ds_load_b128 per lane (layouts
// chosen so each lane's 16 bf16 elements are two contiguous 16B runs in LDS).
typedef __bf16 bf16_t;
typedef __attribute__((ext_vector_type(16))) __bf16 v16bf;
typedef __attribute__((ext_vector_type(8)))  float  v8f;
typedef __attribute__((ext_vector_type(4)))  unsigned int v4u;

#define EPS 1e-5f

union FragU  { v16bf v; v4u q[2]; };          // 32B fragment <-> two 16B words
union Pack8  { v4u q; bf16_t h[8]; };         // 8 bf16 <-> 16B word
union Short8 { v4u q; short s[8]; };          // 8 i16  <-> 16B word

static __device__ inline v8f wmma_bf16(v16bf a, v16bf b, v8f c) {
  return __builtin_amdgcn_wmma_f32_16x16x32_bf16(false, a, false, b, (short)0, c,
                                                 false, false);
}

// A fragment from row-major LDS tile [16 m][lda k] (ISA 7.12.2):
// lane m = lane&15; K runs {koff..koff+7, 16+koff..16+koff+7}, koff=8*(lane>>4)
static __device__ inline v16bf ld_a_frag(const bf16_t* As, int lda) {
  const int lane = threadIdx.x & 31;
  const bf16_t* p = As + (lane & 15) * lda + ((lane >> 4) << 3);
  FragU f;
  f.q[0] = *(const v4u*)p;
  f.q[1] = *(const v4u*)(p + 16);
  return f.v;
}

// B fragment from n-major LDS tile [n][ldk k]:
// lane n = lane&15; K runs {koff..koff+7, 16+koff..16+koff+7}, koff=8*(lane>>4)
static __device__ inline v16bf ld_b_frag_nk(const bf16_t* Bs, int ldk) {
  const int lane = threadIdx.x & 31;
  const bf16_t* p = Bs + (lane & 15) * ldk + ((lane >> 4) << 3);
  FragU f;
  f.q[0] = *(const v4u*)p;
  f.q[1] = *(const v4u*)(p + 16);
  return f.v;
}

// ---------------- weight repack to bf16, n-major [N][Kpad] -------------------
__global__ void prep_weights(const float* __restrict__ c1w,  // [16][1][5][5]
                             const float* __restrict__ c2w,  // [32][16][5][5]
                             const float* __restrict__ f1w,  // [128][512]
                             bf16_t* __restrict__ w1,        // [16][32]
                             bf16_t* __restrict__ w2,        // [32][416]
                             bf16_t* __restrict__ w3) {      // [128][512]
  const int tid = blockIdx.x * blockDim.x + threadIdx.x;
  const int nt = gridDim.x * blockDim.x;
  for (int i = tid; i < 16 * 32; i += nt) {
    int n = i >> 5, k = i & 31;
    w1[i] = (bf16_t)((k < 25) ? c1w[n * 25 + k] : 0.f);
  }
  for (int i = tid; i < 32 * 416; i += nt) {
    int n = i / 416, k = i % 416;
    w2[i] = (bf16_t)((k < 400) ? c2w[n * 400 + k] : 0.f);
  }
  for (int i = tid; i < 128 * 512; i += nt)
    w3[i] = (bf16_t)f1w[i];          // fc1_w is already [N][K]
}

// ------------- conv1 (implicit GEMM M=576,K=25->32,N=16) + ReLU + pool -------
__global__ void __launch_bounds__(256)
conv1_pool(const float* __restrict__ x,      // [B,1,28,28]
           const bf16_t* __restrict__ w1,    // [16][32]
           const float* __restrict__ b1,     // [16]
           bf16_t* __restrict__ act1) {      // [B,16,12,12]
  __shared__ __align__(16) bf16_t img[28 * 28];
  __shared__ __align__(16) bf16_t w1s[16 * 32];
  __shared__ float b1s[16];
  __shared__ float conv[576 * 16];   // [pixel][ch] f32 pre-pool stage
  const int b = blockIdx.x;
  const int tid = threadIdx.x;

  for (int i = tid; i < 784; i += 256) img[i] = (bf16_t)x[b * 784 + i];
  for (int i = tid; i < 512; i += 256) w1s[i] = w1[i];
  if (tid < 16) b1s[tid] = b1[tid];
  __syncthreads();

  const int wave = tid >> 5;
  const int lane = tid & 31;
  const int m = lane & 15;
  const int koff = (lane >> 4) << 3;
  const int nn = lane & 15;
  const int mb = (lane >> 4) << 3;

  // per-lane register table of im2col deltas (loop-invariant; divides hoisted)
  int tbl[16];
#pragma unroll
  for (int e = 0; e < 16; ++e) {
    int k = ((e >> 3) << 4) + koff + (e & 7);
    tbl[e] = (k < 25) ? ((k / 5) * 28 + (k % 5)) : -1;
  }
  const v16bf bw = ld_b_frag_nk(w1s, 32);      // weights: 1 frag, hoisted

  for (int t = wave; t < 36; t += 8) {          // 36 M-tiles of 16 pixels
    const int p = t * 16 + m;
    const int pbase = (p / 24) * 28 + (p % 24); // top-left of 5x5 window
    v16bf a;
#pragma unroll
    for (int e = 0; e < 16; ++e)
      a[e] = (tbl[e] >= 0) ? img[tbl[e] + pbase] : (bf16_t)0.f;
    v8f acc = {};
    acc = wmma_bf16(a, bw, acc);
    const float bias = b1s[nn];
#pragma unroll
    for (int r = 0; r < 8; ++r)
      conv[(t * 16 + mb + r) * 16 + nn] = fmaxf(acc[r] + bias, 0.f);
  }
  __syncthreads();
  for (int i = tid; i < 2304; i += 256) {        // 2x2 maxpool -> [16,12,12]
    int c = i / 144, rem = i % 144;
    int py = rem / 12, px = rem % 12;
    int y = py * 2, xx = px * 2;
    float v0 = conv[(y * 24 + xx) * 16 + c];
    float v1 = conv[(y * 24 + xx + 1) * 16 + c];
    float v2 = conv[((y + 1) * 24 + xx) * 16 + c];
    float v3 = conv[((y + 1) * 24 + xx + 1) * 16 + c];
    act1[b * 2304 + i] = (bf16_t)fmaxf(fmaxf(v0, v1), fmaxf(v2, v3));
  }
}

// ------------- conv2 (implicit GEMM M=64,K=400->416,N=32) + ReLU + pool ------
__global__ void __launch_bounds__(256)
conv2_pool(const bf16_t* __restrict__ act1,   // [B,16,12,12]
           const bf16_t* __restrict__ w2,     // [32][416] n-major
           const float* __restrict__ b2,      // [32]
           bf16_t* __restrict__ act2) {       // [B,512] (= [32,4,4] flattened)
  __shared__ __align__(16) bf16_t a1[16 * 144];
  __shared__ __align__(16) bf16_t w2s[32 * 416];
  __shared__ __align__(16) short  tbl[416];    // k -> c*144 + r*12 + s
  __shared__ __align__(16) bf16_t As[64 * 32]; // staged im2col A-tile
  __shared__ float b2s[32];
  __shared__ float conv[64 * 32];              // [pixel][ch]
  const int b = blockIdx.x;
  const int tid = threadIdx.x;

  {  // 128-bit cooperative LDS fills
    const v4u* s1 = (const v4u*)(act1 + b * 2304);
    v4u* d1 = (v4u*)a1;
    for (int i = tid; i < 288; i += 256) d1[i] = s1[i];
    const v4u* s2 = (const v4u*)w2;            // hot in L2 across all blocks
    v4u* d2 = (v4u*)w2s;
    for (int i = tid; i < 1664; i += 256) d2[i] = s2[i];
    for (int i = tid; i < 416; i += 256) {
      int c = i / 25, rem = i % 25;
      tbl[i] = (i < 400) ? (short)(c * 144 + (rem / 5) * 12 + (rem % 5)) : (short)0;
    }
    if (tid < 32) b2s[tid] = b2[tid];
  }

  const int wave = tid >> 5;
  const int lane = tid & 31;
  const int Mt = wave >> 1, Nt = wave & 1;     // 4 M-tiles x 2 N-tiles, 1/wave

  // staging role: thread -> (row mrow of 64, 8-wide k-chunk)
  const int mrow = tid >> 2;
  const int chunk = tid & 3;
  const int pbase = (mrow >> 3) * 12 + (mrow & 7);
  bf16_t* as_dst = As + mrow * 32 + chunk * 8;

  v8f acc = {};
#pragma unroll 1
  for (int kt = 0; kt < 13; ++kt) {            // K = 13 * 32 = 416
    __syncthreads();                            // frag reads of prev As done
    {   // build 64x32 A-tile cooperatively: table-driven gather, packed store
      const int k0 = kt * 32 + chunk * 8;
      Short8 ts; ts.q = *(const v4u*)&tbl[k0];
      Pack8 pv;
#pragma unroll
      for (int j = 0; j < 8; ++j) {
        int k = k0 + j;
        pv.h[j] = (k < 400) ? a1[(int)ts.s[j] + pbase] : (bf16_t)0.f;
      }
      *(v4u*)as_dst = pv.q;
    }
    __syncthreads();
    v16bf a = ld_a_frag(As + Mt * 16 * 32, 32);
    v16bf bb = ld_b_frag_nk(w2s + (Nt * 16) * 416 + kt * 32, 416);
    acc = wmma_bf16(a, bb, acc);
  }
  const int nn = Nt * 16 + (lane & 15);
  const int mb = (lane >> 4) << 3;
  const float bias = b2s[nn];
#pragma unroll
  for (int r = 0; r < 8; ++r)
    conv[(Mt * 16 + mb + r) * 32 + nn] = fmaxf(acc[r] + bias, 0.f);
  __syncthreads();
  for (int i = tid; i < 512; i += 256) {       // pool -> flat c*16 + py*4 + px
    int c = i >> 4, py2 = (i >> 2) & 3, px2 = i & 3;
    int y = py2 * 2, xx = px2 * 2;
    float v0 = conv[(y * 8 + xx) * 32 + c];
    float v1 = conv[(y * 8 + xx + 1) * 32 + c];
    float v2 = conv[((y + 1) * 8 + xx) * 32 + c];
    float v3 = conv[((y + 1) * 8 + xx + 1) * 32 + c];
    act2[b * 512 + i] = (bf16_t)fmaxf(fmaxf(v0, v1), fmaxf(v2, v3));
  }
}

// ------------------------- fc1: [8192,512] x [512,128]^T ---------------------
__global__ void __launch_bounds__(256)
fc1_gemm(const bf16_t* __restrict__ act2,   // [8192,512]
         const bf16_t* __restrict__ w3,     // [128][512] n-major
         const float* __restrict__ b3,      // [128]
         bf16_t* __restrict__ act3) {       // [8192,128]
  __shared__ __align__(16) bf16_t As[128 * 32];   // [m][k]
  __shared__ __align__(16) bf16_t Bs[128 * 32];   // [n][k]
  __shared__ float b3s[128];
  const int tid = threadIdx.x;
  const int row0 = blockIdx.x * 128;
  if (tid < 128) b3s[tid] = b3[tid];
  const int wave = tid >> 5;
  const int lane = tid & 31;
  v8f acc[8];
#pragma unroll
  for (int i = 0; i < 8; ++i) acc[i] = (v8f){};

#pragma unroll 1
  for (int kt = 0; kt < 16; ++kt) {            // K = 16 * 32 = 512
    __syncthreads();
    for (int v = tid; v < 512; v += 256) {     // A: 128 rows x 32 k, 16B loads
      int r = v >> 2, kk = (v & 3) << 3;
      *(v4u*)&As[r * 32 + kk] =
          *(const v4u*)&act2[(row0 + r) * 512 + kt * 32 + kk];
    }
    for (int v = tid; v < 512; v += 256) {     // B: 128 n x 32 k (n-major)
      int n = v >> 2, kk = (v & 3) << 3;
      *(v4u*)&Bs[n * 32 + kk] = *(const v4u*)&w3[n * 512 + kt * 32 + kk];
    }
    __syncthreads();
    v16bf a = ld_a_frag(As + wave * 16 * 32, 32);   // A reused across 8 Nt
#pragma unroll
    for (int nt = 0; nt < 8; ++nt) {
      v16bf bb = ld_b_frag_nk(Bs + nt * 16 * 32, 32);
      acc[nt] = wmma_bf16(a, bb, acc[nt]);
    }
  }
  const int n0 = lane & 15;
  const int mb = (lane >> 4) << 3;
#pragma unroll
  for (int nt = 0; nt < 8; ++nt) {
#pragma unroll
    for (int r = 0; r < 8; ++r) {
      int row = row0 + wave * 16 + mb + r;
      int col = nt * 16 + n0;
      act3[row * 128 + col] = (bf16_t)fmaxf(acc[nt][r] + b3s[col], 0.f);
    }
  }
}

// -------- fc2 (128->10) + eps-lifted simplex projection + log, f32 VALU ------
__global__ void __launch_bounds__(256)
fc2_project(const bf16_t* __restrict__ act3,  // [8192,128]
            const float* __restrict__ w4,     // [10,128]
            const float* __restrict__ b4,     // [10]
            float* __restrict__ out) {        // [8192,10]
  __shared__ float wsh[10 * 128];
  __shared__ float bs[10];
  const int tid = threadIdx.x;
  for (int i = tid; i < 1280; i += 256) wsh[i] = w4[i];
  if (tid < 10) bs[tid] = b4[tid];
  __syncthreads();

  const int row = blockIdx.x * 256 + tid;
  float acc[10];
#pragma unroll
  for (int j = 0; j < 10; ++j) acc[j] = bs[j];
  const bf16_t* ar = act3 + row * 128;
#pragma unroll 4
  for (int k = 0; k < 128; ++k) {
    float av = (float)ar[k];
#pragma unroll
    for (int j = 0; j < 10; ++j) acc[j] += av * wsh[j * 128 + k];
  }

  // min 0.25||z||^2 - x.z on {z >= EPS, sum z = 1}:
  // v = 2x - EPS projected onto {w>=0, sum w = 1 - 10*EPS}, z = w + EPS
  const float s = 1.0f - 10.0f * EPS;
  float v[10], u[10];
#pragma unroll
  for (int j = 0; j < 10; ++j) { v[j] = 2.0f * acc[j] - EPS; u[j] = v[j]; }
#pragma unroll
  for (int i = 0; i < 9; ++i)
#pragma unroll
    for (int j = 0; j < 9; ++j)
      if (j < 9 - i)
        if (u[j] < u[j + 1]) { float t = u[j]; u[j] = u[j + 1]; u[j + 1] = t; }
  float css = 0.f, css_rho = 0.f;
  int rho = 1;
#pragma unroll
  for (int k = 1; k <= 10; ++k) {
    css += u[k - 1];
    if (u[k - 1] * (float)k > css - s) { rho = k; css_rho = css; }
  }
  const float tau = (css_rho - s) / (float)rho;
#pragma unroll
  for (int j = 0; j < 10; ++j)
    out[row * 10 + j] = logf(fmaxf(v[j] - tau, 0.f) + EPS);
}

// ---------------------------------- launch -----------------------------------
extern "C" void kernel_launch(void* const* d_in, const int* in_sizes, int n_in,
                              void* d_out, int out_size, void* d_ws, size_t ws_size,
                              hipStream_t stream) {
  const float* data = (const float*)d_in[0];
  const float* c1w  = (const float*)d_in[1];
  const float* c1b  = (const float*)d_in[2];
  const float* c2w  = (const float*)d_in[3];
  const float* c2b  = (const float*)d_in[4];
  const float* f1w  = (const float*)d_in[5];
  const float* f1b  = (const float*)d_in[6];
  const float* f2w  = (const float*)d_in[7];
  const float* f2b  = (const float*)d_in[8];
  float* out = (float*)d_out;

  char* ws = (char*)d_ws;                       // ~48.4 MB used
  bf16_t* w1   = (bf16_t*)(ws + 0);             //   1 KB  [16][32]
  bf16_t* w2   = (bf16_t*)(ws + 1024);          //  26 KB  [32][416]
  bf16_t* w3   = (bf16_t*)(ws + 27648);         // 128 KB  [128][512]
  bf16_t* act1 = (bf16_t*)(ws + 158720);        //  36 MB  [8192,16,12,12]
  bf16_t* act2 = (bf16_t*)(ws + 37907456);      //   8 MB  [8192,512]
  bf16_t* act3 = (bf16_t*)(ws + 46296064);      //   2 MB  [8192,128]

  prep_weights<<<64, 256, 0, stream>>>(c1w, c2w, f1w, w1, w2, w3);
  conv1_pool<<<8192, 256, 0, stream>>>(data, w1, c1b, act1);
  conv2_pool<<<8192, 256, 0, stream>>>(act1, w2, c2b, act2);
  fc1_gemm<<<64, 256, 0, stream>>>(act2, w3, f1b, act3);
  fc2_project<<<32, 256, 0, stream>>>(act3, f2w, f2b, out);
}
